// MultiHeadAttention_59330678227182
// MI455X (gfx1250) — matrix operand order repaired and checked by reading, not compiled, not run on hardware
//
#include <hip/hip_runtime.h>

// Problem constants (from reference)
#define BATCH   4
#define S_LEN   2048
#define D_DIM   1024
#define N_HEADS 16
#define D_HEAD  64

typedef __attribute__((ext_vector_type(16))) _Float16 v16h;
typedef __attribute__((ext_vector_type(8)))  float    v8f;
typedef __attribute__((ext_vector_type(4)))  _Float16 v4h;

union FragH {           // one WMMA 16-bit A/B operand: 8 VGPRs = 16 halves
    v16h  v;
    uint4 u[2];
};
union CvH {
    unsigned int u;
    _Float16     h[2];
};

static __device__ __forceinline__ v8f wmma_f16(const FragH& a, const FragH& b, v8f c) {
    // D = A(16x32) * B(32x16) + C, fp32 accumulate
    return __builtin_amdgcn_wmma_f32_16x16x32_f16(
        /*neg_a=*/false, a.v, /*neg_b=*/false, b.v,
        /*c_mod=*/(short)0, c, /*reuse_a=*/false, /*reuse_b=*/false);
}

// ---------------------------------------------------------------------------
// fp32 -> fp16 conversion (vectorized, 4 elems/thread)
// ---------------------------------------------------------------------------
__global__ __launch_bounds__(256) void cvt_f32_f16(const float* __restrict__ src,
                                                   _Float16* __restrict__ dst, int n4) {
    int i = blockIdx.x * 256 + threadIdx.x;
    if (i >= n4) return;
    float4 f = ((const float4*)src)[i];
    v4h h = { (_Float16)f.x, (_Float16)f.y, (_Float16)f.z, (_Float16)f.w };
    ((v4h*)dst)[i] = h;
}

// ---------------------------------------------------------------------------
// Tiled f16 GEMM, fp32 accumulate:  C[M,1024] = A[M,1024] @ B[1024,1024]
// Block tile 128x64, 8 waves, each wave 32x32 (4 WMMA accumulators), BK=32.
// MODE 0: write f16 head-split [B,H,S,DH]   (QKV projections)
// MODE 1: write f32 row-major  [M,N]        (output projection -> d_out)
// Within one block, batch index (m0>>11) and head index (n0>>6) are constant.
// ---------------------------------------------------------------------------
template <int MODE>
__global__ __launch_bounds__(256) void gemm_f16_kernel(
    const _Float16* __restrict__ A, const _Float16* __restrict__ Bw,
    void* __restrict__ outp)
{
    __shared__ _Float16 sA[128 * 32];   // row-major, stride 32
    __shared__ _Float16 sBt[64 * 32];   // column-contiguous: sBt[n*32 + k]

    const int tid  = threadIdx.x;
    const int lane = tid & 31;
    const int wid  = tid >> 5;
    const int r    = lane & 15;
    const int half = lane >> 4;
    const int m0   = blockIdx.y * 128;
    const int n0   = blockIdx.x * 64;
    const int wm   = (wid >> 1) * 32;    // wave row offset inside block tile
    const int wn   = (wid & 1) * 32;     // wave col offset inside block tile

    v8f acc[2][2];
#pragma unroll
    for (int i = 0; i < 2; ++i)
#pragma unroll
        for (int j = 0; j < 2; ++j) acc[i][j] = (v8f)0.0f;

    for (int kt = 0; kt < D_DIM; kt += 32) {
        // ---- stage A tile (128x32), 512 uint4 total, 2 per thread ----
#pragma unroll
        for (int i = 0; i < 2; ++i) {
            int vec = tid + i * 256;
            int row = vec >> 2;
            int c   = (vec & 3) * 8;
            *(uint4*)&sA[row * 32 + c] =
                *(const uint4*)&A[(size_t)(m0 + row) * D_DIM + kt + c];
        }
        // ---- stage B tile transposed (32x64 -> sBt[n][k]), 1024 uints ----
#pragma unroll
        for (int i = 0; i < 4; ++i) {
            int idx = tid + i * 256;
            int k   = idx >> 5;
            int n2  = (idx & 31) * 2;
            CvH cv;
            cv.u = *(const unsigned int*)&Bw[(size_t)(kt + k) * D_DIM + n0 + n2];
            sBt[n2 * 32 + k]       = cv.h[0];
            sBt[(n2 + 1) * 32 + k] = cv.h[1];
        }
        __syncthreads();

        // ---- fragments + 4 WMMAs ----
        FragH a[2], b[2];
#pragma unroll
        for (int i = 0; i < 2; ++i) {
            int base = (wm + i * 16 + r) * 32;
            a[i].u[0] = *(const uint4*)&sA[base + 8 * half];
            a[i].u[1] = *(const uint4*)&sA[base + 16 + 8 * half];
        }
#pragma unroll
        for (int j = 0; j < 2; ++j) {
            int base = (wn + j * 16 + r) * 32;
            b[j].u[0] = *(const uint4*)&sBt[base + 8 * half];
            b[j].u[1] = *(const uint4*)&sBt[base + 16 + 8 * half];
        }
#pragma unroll
        for (int i = 0; i < 2; ++i)
#pragma unroll
            for (int j = 0; j < 2; ++j)
                acc[i][j] = wmma_f16(a[i], b[j], acc[i][j]);
        __syncthreads();
    }

    // ---- epilogue: C layout row = j + 8*half, col = lane&15 ----
    if (MODE == 0) {
        // head-split f16: addr = ((b*H + h)*S + s)*DH + dh
        const int b_ = m0 >> 11;            // block-constant (128 | 2048)
        const int h_ = n0 >> 6;             // block-constant (64 == DH)
        _Float16* ob = (_Float16*)outp +
                       ((size_t)(b_ * N_HEADS + h_)) * S_LEN * D_HEAD;
#pragma unroll
        for (int i = 0; i < 2; ++i) {
            int srow = (m0 & (S_LEN - 1)) + wm + i * 16 + 8 * half;
#pragma unroll
            for (int jj = 0; jj < 2; ++jj) {
                _Float16* p = ob + (size_t)srow * D_HEAD + wn + jj * 16 + r;
#pragma unroll
                for (int j = 0; j < 8; ++j)
                    p[(size_t)j * D_HEAD] = (_Float16)acc[i][jj][j];
            }
        }
    } else {
        // plain f32 row-major [M,1024]
        float* ob = (float*)outp;
#pragma unroll
        for (int i = 0; i < 2; ++i) {
            int row = m0 + wm + i * 16 + 8 * half;
#pragma unroll
            for (int jj = 0; jj < 2; ++jj) {
                float* p = ob + (size_t)row * D_DIM + n0 + wn + jj * 16 + r;
#pragma unroll
                for (int j = 0; j < 8; ++j)
                    p[(size_t)j * D_DIM] = acc[i][jj][j];
            }
        }
    }
}

// ---------------------------------------------------------------------------
// Flash attention: q,k,v f16 [B*H, S, DH]; out f16 [B, S, H*DH].
// One block = (b, h, 128 query rows); 8 waves, wave owns 16 query rows.
// Key loop: 32 blocks of 64 keys. Masked keys get score 1e-9 (NOT -inf),
// matching the reference's sequence_mask-before-softmax semantics.
// ---------------------------------------------------------------------------
__global__ __launch_bounds__(256) void flash_attn_kernel(
    const _Float16* __restrict__ q, const _Float16* __restrict__ k,
    const _Float16* __restrict__ v, const int* __restrict__ valid_lens,
    _Float16* __restrict__ out)
{
    __shared__ _Float16 sK[64 * 64];        // [key][dh]  (column-contig for k^T B-frags)
    __shared__ _Float16 sVt[64 * 64];       // [dh][key]  (column-contig for V B-frags)
    __shared__ _Float16 sP[8][16 * 64];     // per-wave P tile [row][col]

    const int tid  = threadIdx.x;
    const int lane = tid & 31;
    const int wid  = tid >> 5;
    const int r    = lane & 15;
    const int half = lane >> 4;

    const int qb = blockIdx.x & 15;             // query block (128 rows each)
    const int bh = blockIdx.x >> 4;             // batch*heads index
    const int b_ = bh >> 4;
    const int h_ = bh & (N_HEADS - 1);
    const int vl = valid_lens[b_];
    const int q0 = qb * 128 + wid * 16;

    // preload q A-frags (rows fixed for whole kernel)
    FragH aq[2];
    {
        const _Float16* qrow = q + ((size_t)bh * S_LEN + q0 + r) * D_HEAD;
#pragma unroll
        for (int kk = 0; kk < 2; ++kk) {
            aq[kk].u[0] = *(const uint4*)&qrow[32 * kk + 8 * half];
            aq[kk].u[1] = *(const uint4*)&qrow[32 * kk + 16 + 8 * half];
        }
    }

    float mrow[8], lrow[8];
    v8f oacc[4];
#pragma unroll
    for (int j = 0; j < 8; ++j) { mrow[j] = -1e30f; lrow[j] = 0.0f; }
#pragma unroll
    for (int t = 0; t < 4; ++t) oacc[t] = (v8f)0.0f;

    for (int kb = 0; kb < S_LEN / 64; ++kb) {
        // ---- stage K tile [64 keys][64 dh] row-major ----
#pragma unroll
        for (int i = 0; i < 2; ++i) {
            int vec = tid + i * 256;
            int key = vec >> 3;
            int c   = (vec & 7) * 8;
            *(uint4*)&sK[key * 64 + c] =
                *(const uint4*)&k[((size_t)bh * S_LEN + kb * 64 + key) * D_HEAD + c];
        }
        // ---- stage V tile transposed -> sVt[dh][key] ----
#pragma unroll
        for (int i = 0; i < 8; ++i) {
            int idx = tid + i * 256;
            int key = idx >> 5;
            int d2  = (idx & 31) * 2;
            CvH cv;
            cv.u = *(const unsigned int*)&v[((size_t)bh * S_LEN + kb * 64 + key) * D_HEAD + d2];
            sVt[d2 * 64 + key]       = cv.h[0];
            sVt[(d2 + 1) * 64 + key] = cv.h[1];
        }
        __syncthreads();

        // ---- scores: 4 col-tiles of 16 keys, K-dim = DH = 64 (2 wmma steps) ----
        float sf[4][8];
#pragma unroll
        for (int t = 0; t < 4; ++t) {
            v8f acc = (v8f)0.0f;
#pragma unroll
            for (int kk = 0; kk < 2; ++kk) {
                FragH bk;
                int base = (t * 16 + r) * 64 + 32 * kk;
                bk.u[0] = *(const uint4*)&sK[base + 8 * half];
                bk.u[1] = *(const uint4*)&sK[base + 16 + 8 * half];
                acc = wmma_f16(aq[kk], bk, acc);
            }
            int  key_g = kb * 64 + t * 16 + r;
            bool msk   = key_g >= vl;
#pragma unroll
            for (int j = 0; j < 8; ++j)
                sf[t][j] = msk ? 1e-9f : acc[j] * 0.125f;   // 1/sqrt(64)
        }

        // ---- online softmax update ----
        float mnew[8], alpha[8], rsum[8];
#pragma unroll
        for (int j = 0; j < 8; ++j) {
            float mx = fmaxf(fmaxf(sf[0][j], sf[1][j]), fmaxf(sf[2][j], sf[3][j]));
#pragma unroll
            for (int d = 1; d < 16; d <<= 1) mx = fmaxf(mx, __shfl_xor(mx, d, 32));
            mnew[j]  = fmaxf(mrow[j], mx);
            alpha[j] = __expf(mrow[j] - mnew[j]);
            mrow[j]  = mnew[j];
            rsum[j]  = 0.0f;
        }
#pragma unroll
        for (int t = 0; t < 4; ++t)
#pragma unroll
            for (int j = 0; j < 8; ++j) {
                float p = __expf(sf[t][j] - mnew[j]);
                rsum[j] += p;
                sP[wid][(j + 8 * half) * 64 + t * 16 + r] = (_Float16)p;
            }
#pragma unroll
        for (int j = 0; j < 8; ++j) {
            float s = rsum[j];
#pragma unroll
            for (int d = 1; d < 16; d <<= 1) s += __shfl_xor(s, d, 32);
            lrow[j] = lrow[j] * alpha[j] + s;
        }
#pragma unroll
        for (int t = 0; t < 4; ++t)
#pragma unroll
            for (int j = 0; j < 8; ++j) oacc[t][j] *= alpha[j];

        // ---- O += P(16x64) @ V(64x64): 2 K-steps x 4 dh tiles ----
#pragma unroll
        for (int kk = 0; kk < 2; ++kk) {
            FragH ap;
            int base = r * 64 + 32 * kk;
            ap.u[0] = *(const uint4*)&sP[wid][base + 8 * half];
            ap.u[1] = *(const uint4*)&sP[wid][base + 16 + 8 * half];
#pragma unroll
            for (int t = 0; t < 4; ++t) {
                FragH bv;
                int vb = (t * 16 + r) * 64 + 32 * kk;
                bv.u[0] = *(const uint4*)&sVt[vb + 8 * half];
                bv.u[1] = *(const uint4*)&sVt[vb + 16 + 8 * half];
                oacc[t] = wmma_f16(ap, bv, oacc[t]);
            }
        }
        __syncthreads();
    }

    // ---- finalize: O / l, write [B, S, H*DH] f16; base ptr + j*D_DIM ----
    {
        int sg0 = q0 + 8 * half;
        _Float16* ob = out + ((size_t)(b_ * S_LEN + sg0)) * D_DIM + h_ * D_HEAD + r;
#pragma unroll
        for (int t = 0; t < 4; ++t) {
            _Float16* p = ob + t * 16;
#pragma unroll
            for (int j = 0; j < 8; ++j)
                p[(size_t)j * D_DIM] = (_Float16)(oacc[t][j] / lrow[j]);
        }
    }
}

// ---------------------------------------------------------------------------
// Host launch
// ---------------------------------------------------------------------------
extern "C" void kernel_launch(void* const* d_in, const int* in_sizes, int n_in,
                              void* d_out, int out_size, void* d_ws, size_t ws_size,
                              hipStream_t stream) {
    (void)in_sizes; (void)n_in; (void)out_size; (void)ws_size;

    const float* Q  = (const float*)d_in[0];
    const float* K  = (const float*)d_in[1];
    const float* V  = (const float*)d_in[2];
    const int*   VL = (const int*)d_in[3];
    const float* Wq = (const float*)d_in[4];
    const float* Wk = (const float*)d_in[5];
    const float* Wv = (const float*)d_in[6];
    const float* Wo = (const float*)d_in[7];

    const size_t nX  = (size_t)BATCH * S_LEN * D_DIM;      // 8,388,608
    const size_t nW  = (size_t)D_DIM * D_DIM;              // 1,048,576
    const size_t szX = nX * sizeof(_Float16);              // 16 MiB
    const size_t szW = nW * sizeof(_Float16);              // 2 MiB

    char* ws = (char*)d_ws;
    _Float16* Qh  = (_Float16*)(ws);
    _Float16* Kh  = (_Float16*)(ws + szX);
    _Float16* Vh  = (_Float16*)(ws + 2 * szX);
    _Float16* Wqh = (_Float16*)(ws + 3 * szX);
    _Float16* Wkh = (_Float16*)(ws + 3 * szX + szW);
    _Float16* Wvh = (_Float16*)(ws + 3 * szX + 2 * szW);
    _Float16* Woh = (_Float16*)(ws + 3 * szX + 3 * szW);
    _Float16* qp  = (_Float16*)(ws + 3 * szX + 4 * szW);           // [B,H,S,DH]
    _Float16* kp  = (_Float16*)(ws + 4 * szX + 4 * szW);
    _Float16* vp  = (_Float16*)(ws + 5 * szX + 4 * szW);
    _Float16* ao  = (_Float16*)(ws + 6 * szX + 4 * szW);           // [B,S,D]

    // 1) fp32 -> fp16 conversions
    const int nX4 = (int)(nX / 4), nW4 = (int)(nW / 4);
    cvt_f32_f16<<<(nX4 + 255) / 256, 256, 0, stream>>>(Q, Qh, nX4);
    cvt_f32_f16<<<(nX4 + 255) / 256, 256, 0, stream>>>(K, Kh, nX4);
    cvt_f32_f16<<<(nX4 + 255) / 256, 256, 0, stream>>>(V, Vh, nX4);
    cvt_f32_f16<<<(nW4 + 255) / 256, 256, 0, stream>>>(Wq, Wqh, nW4);
    cvt_f32_f16<<<(nW4 + 255) / 256, 256, 0, stream>>>(Wk, Wkh, nW4);
    cvt_f32_f16<<<(nW4 + 255) / 256, 256, 0, stream>>>(Wv, Wvh, nW4);
    cvt_f32_f16<<<(nW4 + 255) / 256, 256, 0, stream>>>(Wo, Woh, nW4);

    // 2) QKV projections: [8192,1024]@[1024,1024] -> head-split f16
    dim3 ggrid(D_DIM / 64, (BATCH * S_LEN) / 128);   // (16, 64)
    gemm_f16_kernel<0><<<ggrid, 256, 0, stream>>>(Qh, Wqh, qp);
    gemm_f16_kernel<0><<<ggrid, 256, 0, stream>>>(Kh, Wkh, kp);
    gemm_f16_kernel<0><<<ggrid, 256, 0, stream>>>(Vh, Wvh, vp);

    // 3) flash attention: grid = B*H * (S/128) = 1024 blocks
    flash_attn_kernel<<<BATCH * N_HEADS * (S_LEN / 128), 256, 0, stream>>>(
        qp, kp, vp, VL, ao);

    // 4) output projection -> fp32 d_out
    gemm_f16_kernel<1><<<ggrid, 256, 0, stream>>>(ao, Woh, (void*)d_out);
}
